// GIN_47674136985670
// MI455X (gfx1250) — matrix operand order repaired and checked by reading, not compiled, not run on hardware
//
#include <hip/hip_runtime.h>

typedef __attribute__((ext_vector_type(16))) __bf16 v16bf;
typedef __attribute__((ext_vector_type(8)))  float  v8f;
typedef unsigned int v4u __attribute__((ext_vector_type(4)));
typedef int v4i __attribute__((ext_vector_type(4)));
typedef int v8i __attribute__((ext_vector_type(8)));

#define N_NODES 200000
#define N_EDGES 800000
#define N_MOL   10000
#define IN_DIM  40
#define HDIM    64

#if defined(__HIP_DEVICE_COMPILE__) && __has_builtin(__builtin_amdgcn_tensor_load_to_lds)
#define USE_TDM 1
#else
#define USE_TDM 0
#endif

// WMMA 16x16x32 bf16 operand K patterns (per CDNA5 ISA §7.12.2)
__device__ __forceinline__ int kA(int v, int h) {
    return (v < 4) ? (2 * v + 8 * h) : (16 + 2 * (v - 4) + 8 * h);
}
__device__ __forceinline__ int kB(int v, int h) {
    return (v < 4) ? (2 * v + 8 * h) : (16 + 2 * (v - 4) + 8 * h);
}
__device__ __forceinline__ float leaky(float x) { return x > 0.f ? x : 0.01f * x; }
__device__ __forceinline__ unsigned short f2bfu(float f) {
    return __builtin_bit_cast(unsigned short, (__bf16)f);
}

// ---------------------------------------------------------------- prep
struct PrepArgs {
    const float *W1, *W2, *W3, *Wl1;
    const float *b[3], *mean[3], *var[3], *g[3], *beta[3];
    unsigned short *W1b, *W2b, *W3b, *Wl1b;  // bf16 storage
    float *sf, *tf;                          // 3*64 BN fold scale/shift
};

__global__ void prep_kernel(PrepArgs a) {
    int tid = blockIdx.x * 256 + threadIdx.x;
    if (tid < 65536) { a.Wl1b[tid] = f2bfu(a.Wl1[tid]); return; }
    int t = tid - 65536;
    if (t < 4096) {  // W1: 64x40 -> 64x64 k-padded
        int n = t >> 6, k = t & 63;
        a.W1b[t] = (k < IN_DIM) ? f2bfu(a.W1[n * IN_DIM + k]) : (unsigned short)0;
        return;
    }
    t -= 4096;
    if (t < 4096) { a.W2b[t] = f2bfu(a.W2[t]); return; }
    t -= 4096;
    if (t < 4096) { a.W3b[t] = f2bfu(a.W3[t]); return; }
    t -= 4096;
    if (t < 192) {  // BN fold: out = s*acc + tt, s=g*rsqrt(v+eps), tt=s*(b-m)+beta
        int l = t >> 6, c = t & 63;
        float s = a.g[l][c] * rsqrtf(a.var[l][c] + 1e-5f);
        a.sf[t] = s;
        a.tf[t] = s * (a.b[l][c] - a.mean[l][c]) + a.beta[l][c];
    }
}

// ------------------------------------------------- edge message + scatter
// msg = relu(x[src] + edge_attr @ We^T + be); agg[dst] += msg  (atomic f32)
template <int DIN>
__global__ __launch_bounds__(256) void edge_kernel(
    const float* __restrict__ x, const float* __restrict__ ea,
    const int* __restrict__ src, const int* __restrict__ dst,
    const float* __restrict__ We, const float* __restrict__ be,
    float* __restrict__ agg)
{
    __shared__ float sW[DIN * 16];
    __shared__ float sb[DIN];
    for (int i = threadIdx.x; i < DIN * 16; i += 256) sW[i] = We[i];
    for (int i = threadIdx.x; i < DIN; i += 256) sb[i] = be[i];
    __syncthreads();

    const int OPG = DIN / 4;                      // 4 outputs per thread
    int tid = blockIdx.x * 256 + threadIdx.x;
    int total = N_EDGES * OPG;
    if (tid >= total) return;
    int e = tid / OPG;
    int o4 = tid % OPG;

    const float4* eav = (const float4*)(ea + (size_t)e * 16);
    float4 e0 = eav[0], e1 = eav[1], e2 = eav[2], e3 = eav[3];
    int s = src[e], d = dst[e];
    float4 xs = *(const float4*)(x + (size_t)s * DIN + 4 * o4);
    float xsv[4] = {xs.x, xs.y, xs.z, xs.w};
    float* aggp = agg + (size_t)d * DIN + 4 * o4;

#pragma unroll
    for (int j = 0; j < 4; ++j) {
        int o = 4 * o4 + j;
        const float* w = sW + o * 16;
        float acc = sb[o];
        acc += e0.x * w[0]  + e0.y * w[1]  + e0.z * w[2]  + e0.w * w[3];
        acc += e1.x * w[4]  + e1.y * w[5]  + e1.z * w[6]  + e1.w * w[7];
        acc += e2.x * w[8]  + e2.y * w[9]  + e2.z * w[10] + e2.w * w[11];
        acc += e3.x * w[12] + e3.y * w[13] + e3.z * w[14] + e3.w * w[15];
        float msg = xsv[j] + acc;
        msg = msg > 0.f ? msg : 0.f;
        __hip_atomic_fetch_add(aggp + j, msg, __ATOMIC_RELAXED, __HIP_MEMORY_SCOPE_AGENT);
    }
}

// --------------------------------------- node GEMM: h = BN((x+agg)@W^T+b), leaky
template <int DIN>
__global__ __launch_bounds__(256) void node_gemm_kernel(
    const float* __restrict__ x, const float* __restrict__ agg,
    const unsigned short* __restrict__ Wb,   // bf16 64x64 (k-padded)
    const float* __restrict__ sf, const float* __restrict__ tf,
    float* __restrict__ hout)
{
    const int lane = threadIdx.x & 31;
    const int wave = threadIdx.x >> 5;
    const int row0 = (blockIdx.x * 8 + wave) * 16;
    if (row0 >= N_NODES) return;  // wave-uniform exit (EXEC stays all-ones for WMMA)
    const int h = lane >> 4;
    const int rl = lane & 15;
    const int r = row0 + rl < N_NODES ? row0 + rl : N_NODES - 1;
    const float* xr = x + (size_t)r * DIN;
    const float* ar = agg + (size_t)r * DIN;

    v8f acc[4] = {};
    const unsigned* Wu = (const unsigned*)Wb;
#pragma unroll
    for (int s = 0; s < 2; ++s) {
        v16bf a;
#pragma unroll
        for (int v = 0; v < 8; ++v) {
            int k = s * 32 + kA(v, h);
            float f0 = 0.f, f1 = 0.f;
            if (k < DIN) { f0 = xr[k] + ar[k]; f1 = xr[k + 1] + ar[k + 1]; }
            a[2 * v] = (__bf16)f0;
            a[2 * v + 1] = (__bf16)f1;
        }
#pragma unroll
        for (int nt = 0; nt < 4; ++nt) {
            int n = nt * 16 + rl;
            union { v16bf v; unsigned u[8]; } b;
#pragma unroll
            for (int v = 0; v < 8; ++v) {
                int k = s * 32 + kB(v, h);          // pair (k,k+1), k even
                b.u[v] = Wu[(n * 64 + k) >> 1];     // B[k][n] = W[n][k]
            }
            acc[nt] = __builtin_amdgcn_wmma_f32_16x16x32_bf16(
                false, a, false, b.v, (short)0, acc[nt], false, false);
        }
    }
#pragma unroll
    for (int nt = 0; nt < 4; ++nt) {
        int col = nt * 16 + rl;
        float sc = sf[col], tc = tf[col];
#pragma unroll
        for (int v = 0; v < 8; ++v) {
            int rowm = row0 + v + 8 * h;
            if (rowm < N_NODES)
                hout[(size_t)rowm * 64 + col] = leaky(sc * acc[nt][v] + tc);
        }
    }
}

// ---------------------------------------------------------------- pooling
__global__ void pool_kernel(const float* __restrict__ h3,
                            const int* __restrict__ batch,
                            float* __restrict__ pool)
{
    int tid = blockIdx.x * 256 + threadIdx.x;
    if (tid >= N_NODES * 64) return;
    int n = tid >> 6, c = tid & 63;
    __hip_atomic_fetch_add(pool + (size_t)batch[n] * 64 + c, h3[tid],
                           __ATOMIC_RELAXED, __HIP_MEMORY_SCOPE_AGENT);
}

#if USE_TDM
// Issue a TDM load of one K-phase of Wl1 (bf16): 2-D tile, 256 rows x 64 dwords,
// row stride 128 dwords, packed contiguously into LDS at lds_off.
// D# layout per CDNA5 ISA ch.8 (group0 128b, group1 256b; groups 2/3 zero = 2-D).
__device__ __forceinline__ void tdm_load_wl1(const unsigned short* gsrc, unsigned lds_off) {
    unsigned long long ga = (unsigned long long)gsrc;
    v4u g0;
    g0[0] = 1u;                                   // count=1, user descriptor
    g0[1] = lds_off;                              // lds_addr (bytes)
    g0[2] = (unsigned)(ga & 0xffffffffu);         // global_addr[31:0]
    g0[3] = (unsigned)((ga >> 32) & 0x1ffffffu)   // global_addr[56:32]
          | (2u << 30);                           // type=2 (image)
    v8i g1;
    g1[0] = (int)(2u << 16);                      // wg_mask=0, data_size=2 (4 bytes)
    g1[1] = (int)(128u << 16);                    // tensor_dim0[15:0]=128 (dwords/row)
    g1[2] = (int)(256u << 16);                    // dim0 hi=0 | tensor_dim1[15:0]=256
    g1[3] = (int)(64u << 16);                     // dim1 hi=0 | tile_dim0=64 dwords
    g1[4] = (int)(256u);                          // tile_dim1=256 rows, tile_dim2=0
    g1[5] = (int)(128u);                          // tensor_dim0_stride[31:0]=128 dwords
    g1[6] = 0;                                    // stride hi, dim1_stride lo
    g1[7] = 0;
    v4i z4 = {0, 0, 0, 0};
    v8i z8 = {0, 0, 0, 0, 0, 0, 0, 0};
    // 6-arg form (amdgpu-toolchain clang): (g0, g1, g2, g3, extra, cpol)
    __builtin_amdgcn_tensor_load_to_lds(g0, g1, z4, z4, z8, 0);
}
#endif

// -------------------------------------------------------------- final MLP
// a = [h1|h2|h3|pool[batch]] (K=256); h = leaky(a@Wl1^T+bl1); out = h@Wl2^T+bl2
// Wl1 (bf16) double-buffered in LDS: TDM DMA of phase p+1 overlaps WMMAs of phase p.
__global__ __launch_bounds__(256) void mlp_kernel(
    const float* __restrict__ h1, const float* __restrict__ h2,
    const float* __restrict__ h3, const float* __restrict__ pool,
    const int* __restrict__ batch, const unsigned short* __restrict__ Wl1b,
    const float* __restrict__ bl1, const float* __restrict__ Wl2,
    const float* __restrict__ bl2, float* __restrict__ out)
{
    __shared__ unsigned short sw[2][256 * 128];  // 2 x 64KB K-phase buffers
    const int lane = threadIdx.x & 31;
    const int wave = threadIdx.x >> 5;
    const int row0 = (blockIdx.x * 8 + wave) * 16;
    const int h = lane >> 4;
    const int rl = lane & 15;
    const int r = row0 + rl < N_NODES ? row0 + rl : N_NODES - 1;
    const int mol = batch[r];
    const float* srcs[4] = { h1 + (size_t)r * 64, h2 + (size_t)r * 64,
                             h3 + (size_t)r * 64, pool + (size_t)mol * 64 };

    v8f acc[16] = {};

#if USE_TDM
    // phase 0 DMA, then prefetch phase 1 while computing phase 0
    if (wave == 0) {
        tdm_load_wl1(Wl1b, (unsigned)(unsigned long long)(const void*)&sw[0][0]);
        __builtin_amdgcn_s_wait_tensorcnt(0);
    }
    __syncthreads();
    if (wave == 0)
        tdm_load_wl1(Wl1b + 128, (unsigned)(unsigned long long)(const void*)&sw[1][0]);
#else
    // fallback: synchronous staging of both phases
    {
        unsigned* swu0 = (unsigned*)&sw[0][0];
        const unsigned* gwu = (const unsigned*)Wl1b;
        for (int u = threadIdx.x; u < 32768; u += 256) {
            int p = u >> 14, rem = u & 16383;
            int n = rem >> 6, kp = rem & 63;
            swu0[u] = gwu[n * 128 + p * 64 + kp];
        }
    }
    __syncthreads();
#endif

#pragma unroll
    for (int p = 0; p < 2; ++p) {               // K phase: cols [p*128, p*128+128)
#if USE_TDM
        if (p == 1) {
            if (wave == 0) __builtin_amdgcn_s_wait_tensorcnt(0);
            __syncthreads();                    // publish phase-1 buffer to all waves
        }
#endif
        const unsigned* swu = (const unsigned*)&sw[p][0];
#pragma unroll
        for (int sq = 0; sq < 4; ++sq) {
            int s = p * 4 + sq;                       // global k-step 0..7
            const float* arow = srcs[s >> 1];         // each 32-k block within one src
            v16bf a;
#pragma unroll
            for (int v = 0; v < 8; ++v) {
                int col = (s & 1) * 32 + kA(v, h);
                a[2 * v] = (__bf16)arow[col];
                a[2 * v + 1] = (__bf16)arow[col + 1];
            }
#pragma unroll
            for (int nt = 0; nt < 16; ++nt) {
                int n = nt * 16 + rl;
                union { v16bf v; unsigned u[8]; } b;
#pragma unroll
                for (int v = 0; v < 8; ++v) {
                    int kip = sq * 32 + kB(v, h);     // k within phase, even
                    b.u[v] = swu[n * 64 + (kip >> 1)];
                }
                acc[nt] = __builtin_amdgcn_wmma_f32_16x16x32_bf16(
                    false, a, false, b.v, (short)0, acc[nt], false, false);
            }
        }
    }

    // epilogue: leaky(acc + bl1) dotted with Wl2, reduced across 16 lanes
    float p2[8] = {};
#pragma unroll
    for (int nt = 0; nt < 16; ++nt) {
        int col = nt * 16 + rl;
        float w2 = Wl2[col], bl = bl1[col];
#pragma unroll
        for (int v = 0; v < 8; ++v)
            p2[v] += leaky(acc[nt][v] + bl) * w2;
    }
    float blv = bl2[0];
#pragma unroll
    for (int v = 0; v < 8; ++v) {
        float s = p2[v];
        s += __shfl_xor(s, 1, 16);
        s += __shfl_xor(s, 2, 16);
        s += __shfl_xor(s, 4, 16);
        s += __shfl_xor(s, 8, 16);
        if (rl == v) {
            int rowm = row0 + v + 8 * h;
            if (rowm < N_NODES) out[rowm] = s + blv;
        }
    }
}

// ---------------------------------------------------------------- launch
extern "C" void kernel_launch(void* const* d_in, const int* in_sizes, int n_in,
                              void* d_out, int out_size, void* d_ws, size_t ws_size,
                              hipStream_t stream) {
    (void)in_sizes; (void)n_in; (void)out_size; (void)ws_size;
    const float* x    = (const float*)d_in[0];
    const int*   eidx = (const int*)d_in[1];
    const float* eatt = (const float*)d_in[2];
    const int*   batch = (const int*)d_in[3];
    const float *We1 = (const float*)d_in[4],  *be1 = (const float*)d_in[5];
    const float *W1  = (const float*)d_in[6],  *b1  = (const float*)d_in[7];
    const float *g1  = (const float*)d_in[8],  *bb1 = (const float*)d_in[9];
    const float *m1  = (const float*)d_in[10], *v1  = (const float*)d_in[11];
    const float *We2 = (const float*)d_in[12], *be2 = (const float*)d_in[13];
    const float *W2  = (const float*)d_in[14], *b2  = (const float*)d_in[15];
    const float *g2  = (const float*)d_in[16], *bb2 = (const float*)d_in[17];
    const float *m2  = (const float*)d_in[18], *v2  = (const float*)d_in[19];
    const float *We3 = (const float*)d_in[20], *be3 = (const float*)d_in[21];
    const float *W3  = (const float*)d_in[22], *b3  = (const float*)d_in[23];
    const float *g3  = (const float*)d_in[24], *bb3 = (const float*)d_in[25];
    const float *m3  = (const float*)d_in[26], *v3  = (const float*)d_in[27];
    const float *Wl1 = (const float*)d_in[28], *bl1 = (const float*)d_in[29];
    const float *Wl2 = (const float*)d_in[30], *bl2 = (const float*)d_in[31];
    const int* src = eidx;
    const int* dst = eidx + N_EDGES;

    char* base = (char*)d_ws;
    size_t off = 0;
    auto carve = [&](size_t bytes) -> char* {
        char* p = base + off;
        off = (off + bytes + 255) & ~(size_t)255;
        return p;
    };
    float* h1   = (float*)carve((size_t)N_NODES * 64 * 4);
    float* h2   = (float*)carve((size_t)N_NODES * 64 * 4);
    float* h3   = (float*)carve((size_t)N_NODES * 64 * 4);
    float* agg  = (float*)carve((size_t)N_NODES * 64 * 4);
    float* pool = (float*)carve((size_t)N_MOL * 64 * 4);
    unsigned short* W1b  = (unsigned short*)carve(4096 * 2);
    unsigned short* W2b  = (unsigned short*)carve(4096 * 2);
    unsigned short* W3b  = (unsigned short*)carve(4096 * 2);
    unsigned short* Wl1b = (unsigned short*)carve(65536 * 2);
    float* sf = (float*)carve(192 * 4);
    float* tf = (float*)carve(192 * 4);

    PrepArgs pa = { W1, W2, W3, Wl1,
                    {b1, b2, b3}, {m1, m2, m3}, {v1, v2, v3},
                    {g1, g2, g3}, {bb1, bb2, bb3},
                    W1b, W2b, W3b, Wl1b, sf, tf };
    prep_kernel<<<(65536 + 3 * 4096 + 192 + 255) / 256, 256, 0, stream>>>(pa);

    const int node_blocks = (N_NODES + 127) / 128;  // 8 waves * 16 rows per block

    // layer 1 (din = 40)
    (void)hipMemsetAsync(agg, 0, (size_t)N_NODES * IN_DIM * 4, stream);
    edge_kernel<IN_DIM><<<(N_EDGES * (IN_DIM / 4) + 255) / 256, 256, 0, stream>>>(
        x, eatt, src, dst, We1, be1, agg);
    node_gemm_kernel<IN_DIM><<<node_blocks, 256, 0, stream>>>(
        x, agg, W1b, sf + 0, tf + 0, h1);

    // layer 2 (din = 64)
    (void)hipMemsetAsync(agg, 0, (size_t)N_NODES * 64 * 4, stream);
    edge_kernel<64><<<(N_EDGES * 16 + 255) / 256, 256, 0, stream>>>(
        h1, eatt, src, dst, We2, be2, agg);
    node_gemm_kernel<64><<<node_blocks, 256, 0, stream>>>(
        h1, agg, W2b, sf + 64, tf + 64, h2);

    // layer 3 (din = 64)
    (void)hipMemsetAsync(agg, 0, (size_t)N_NODES * 64 * 4, stream);
    edge_kernel<64><<<(N_EDGES * 16 + 255) / 256, 256, 0, stream>>>(
        h2, eatt, src, dst, We3, be3, agg);
    node_gemm_kernel<64><<<node_blocks, 256, 0, stream>>>(
        h2, agg, W3b, sf + 128, tf + 128, h3);

    // global add pool + final MLP
    (void)hipMemsetAsync(pool, 0, (size_t)N_MOL * 64 * 4, stream);
    pool_kernel<<<(N_NODES * 64 + 255) / 256, 256, 0, stream>>>(h3, batch, pool);
    mlp_kernel<<<node_blocks, 256, 0, stream>>>(
        h1, h2, h3, pool, batch, Wl1b, bl1, Wl2, bl2, (float*)d_out);
}
